// Net_74552042324489
// MI455X (gfx1250) — matrix operand order repaired and checked by reading, not compiled, not run on hardware
//
#include <hip/hip_runtime.h>
#include <hip/hip_bf16.h>
#include <math.h>

typedef __bf16 v16bf __attribute__((ext_vector_type(16)));
typedef float  v8f   __attribute__((ext_vector_type(8)));

#define HW   3136
#define OW   56
#define CQK  192

#if defined(__has_builtin) && __has_builtin(__builtin_amdgcn_sched_barrier)
#define SCHED_FENCE() __builtin_amdgcn_sched_barrier(0)
#else
#define SCHED_FENCE() asm volatile("" ::: "memory")
#endif

__device__ __forceinline__ unsigned short f2bf(float f) {
    unsigned u = __float_as_uint(f);
    u += 0x7FFFu + ((u >> 16) & 1u);   // round-to-nearest-even
    return (unsigned short)(u >> 16);
}

// ---------------- cam = fc8_w (4x512) @ _4 (512xHW) per batch ----------------
__global__ void cam_kernel(const float* __restrict__ w, const float* __restrict__ x4,
                           float* __restrict__ cam) {
    int id = blockIdx.x * blockDim.x + threadIdx.x;     // (n*4+c)*HW + p
    int p = id % HW; int nc = id / HW; int c = nc & 3; int n = nc >> 2;
    const float* wr = w + c * 512;
    const float* xb = x4 + (size_t)n * 512 * HW + p;
    float s = 0.f;
    for (int k = 0; k < 512; ++k) s = fmaf(wr[k], xb[(size_t)k * HW], s);
    cam[id] = s;
}

// ---------------- per-(n,c) min/max normalize + fg/bg logic ----------------
__global__ void norm_kernel(const float* __restrict__ cam, float* __restrict__ camn) {
    __shared__ float smn[4][256];
    __shared__ float smx[4][256];
    int n = blockIdx.x, t = threadIdx.x;
    float mn[4], mx[4];
    #pragma unroll
    for (int c = 0; c < 4; ++c) { mn[c] = 3.4e38f; mx[c] = -3.4e38f; }
    const float* base = cam + (size_t)n * 4 * HW;
    for (int p = t; p < HW; p += 256)
        #pragma unroll
        for (int c = 0; c < 4; ++c) {
            float v = base[c * HW + p];
            mn[c] = fminf(mn[c], v); mx[c] = fmaxf(mx[c], v);
        }
    #pragma unroll
    for (int c = 0; c < 4; ++c) { smn[c][t] = mn[c]; smx[c][t] = mx[c]; }
    __syncthreads();
    for (int s = 128; s > 0; s >>= 1) {
        if (t < s)
            #pragma unroll
            for (int c = 0; c < 4; ++c) {
                smn[c][t] = fminf(smn[c][t], smn[c][t + s]);
                smx[c][t] = fmaxf(smx[c][t], smx[c][t + s]);
            }
        __syncthreads();
    }
    float lmn[4], linv[4];
    #pragma unroll
    for (int c = 0; c < 4; ++c) {
        lmn[c]  = smn[c][0];
        linv[c] = 1.f / (smx[c][0] + 1e-5f - smn[c][0]);
    }
    float* ob = camn + (size_t)n * 4 * HW;
    for (int p = t; p < HW; p += 256) {
        float f1 = (base[1 * HW + p] - lmn[1]) * linv[1];
        float f2 = (base[2 * HW + p] - lmn[2]) * linv[2];
        float f3 = (base[3 * HW + p] - lmn[3]) * linv[3];
        float fm = fmaxf(f1, fmaxf(f2, f3));
        ob[0 * HW + p] = 1.f - fm;
        ob[1 * HW + p] = (f1 < fm) ? 0.f : f1;
        ob[2 * HW + p] = (f2 < fm) ? 0.f : f2;
        ob[3 * HW + p] = (f3 < fm) ? 0.f : f3;
    }
}

// ---------------- bilinear align_corners resize (generic) ----------------
__device__ __forceinline__ float bilin_ac(const float* __restrict__ b, int iw, int ih,
                                          int ox, int oy, float sc) {
    float ys = oy * sc, xs = ox * sc;
    int y0 = (int)ys, x0 = (int)xs;
    float wy = ys - y0, wx = xs - x0;
    int y1 = (y0 + 1 > ih - 1) ? ih - 1 : y0 + 1;
    int x1 = (x0 + 1 > iw - 1) ? iw - 1 : x0 + 1;
    float top = b[y0 * iw + x0] * (1.f - wx) + b[y0 * iw + x1] * wx;
    float bot = b[y1 * iw + x0] * (1.f - wx) + b[y1 * iw + x1] * wx;
    return top * (1.f - wy) + bot * wy;
}

__global__ void resize_x_kernel(const float* __restrict__ x, float* __restrict__ f) {
    int id = blockIdx.x * blockDim.x + threadIdx.x;     // (n*3+c)*HW + p
    int p = id % HW; int nc = id / HW; int c = nc % 3; int n = nc / 3;
    const float* b = x + (size_t)(n * 3 + c) * 448 * 448;
    float v = bilin_ac(b, 448, 448, p % OW, p / OW, 447.f / 55.f);
    f[((size_t)n * 195 + c) * HW + p] = v;
}

__global__ void resize_x2_kernel(const float* __restrict__ x2, float* __restrict__ x2r) {
    int id = blockIdx.x * blockDim.x + threadIdx.x;     // (n*128+c)*HW + p
    int p = id % HW; int nc = id / HW; int c = nc % 128; int n = nc / 128;
    const float* b = x2 + (size_t)(n * 128 + c) * 112 * 112;
    x2r[id] = bilin_ac(b, 112, 112, p % OW, p / OW, 111.f / 55.f);
}

// ---------------- f8_3 = relu(f83_w (64x128) @ x2r) -> f[3..66] ----------------
__global__ void f83_kernel(const float* __restrict__ w, const float* __restrict__ x2r,
                           float* __restrict__ f) {
    int id = blockIdx.x * blockDim.x + threadIdx.x;     // (n*64+o)*HW + p
    int p = id % HW; int no = id / HW; int o = no % 64; int n = no / 64;
    const float* wr = w + o * 128;
    const float* xb = x2r + (size_t)n * 128 * HW + p;
    float s = 0.f;
    for (int c = 0; c < 128; ++c) s = fmaf(wr[c], xb[(size_t)c * HW], s);
    f[((size_t)n * 195 + 3 + o) * HW + p] = fmaxf(s, 0.f);
}

// ---------------- f8_4 = relu(f84_w (128x320) @ deep3) -> f[67..194] ----------------
__global__ void f84_kernel(const float* __restrict__ w, const float* __restrict__ d3,
                           float* __restrict__ f) {
    int id = blockIdx.x * blockDim.x + threadIdx.x;     // (n*128+o)*HW + p
    int p = id % HW; int no = id / HW; int o = no % 128; int n = no / 128;
    const float* wr = w + o * 320;
    const float* xb = d3 + (size_t)n * 320 * HW + p;
    float s = 0.f;
    for (int c = 0; c < 320; ++c) s = fmaf(wr[c], xb[(size_t)c * HW], s);
    f[((size_t)n * 195 + 67 + o) * HW + p] = fmaxf(s, 0.f);
}

// ---------------- q,k projections -> bf16, layout [n][pixel][192] ----------------
__global__ void qk_kernel(const float* __restrict__ wq, const float* __restrict__ wk,
                          const float* __restrict__ f,
                          unsigned short* __restrict__ Qt, unsigned short* __restrict__ Kt) {
    int id = blockIdx.x * blockDim.x + threadIdx.x;     // ((n*HW+p)*192 + o)
    int o = id % CQK; int np = id / CQK; int p = np % HW; int n = np / HW;
    const float* fb  = f + (size_t)n * 195 * HW + p;
    const float* wqr = wq + o * 195;
    const float* wkr = wk + o * 195;
    float sq = 0.f, sk = 0.f;
    for (int c = 0; c < 195; ++c) {
        float v = fb[(size_t)c * HW];
        sq = fmaf(wqr[c], v, sq);
        sk = fmaf(wkr[c], v, sk);
    }
    Qt[id] = f2bf(sq);
    Kt[id] = f2bf(sk);
}

// ---------------- fused attention: S=QtᵀK, online softmax over q, cam @ P ----------------
// grid (49, 8), block 128 (4 waves). Wave w owns 16 k-columns. Column j state lives
// (duplicated) in lanes j and j+16; combine via shfl_xor(16).
// Double-buffered LDS q-tile: one barrier per iteration; staging of tile qt+1
// overlaps the WMMA chain + softmax of tile qt. A sched_barrier between the
// A-tile LDS loads and the WMMA chain keeps all six A octets live so the six
// WMMAs issue back-to-back with graduated dscnt waits.
__global__ void __launch_bounds__(128)
attn_kernel(const unsigned short* __restrict__ Qt, const unsigned short* __restrict__ Kt,
            const float* __restrict__ camn, float* __restrict__ out) {
    __shared__ __align__(16) unsigned short sQ[2][16 * CQK];   // 2 x 6 KB q-tiles
    __shared__ float sCam[2][4 * 16];

    const int n    = blockIdx.y;
    const int wave = threadIdx.x >> 5;
    const int lane = threadIdx.x & 31;
    const int half = lane >> 4;          // 0: rows 0-7 of C/D, 1: rows 8-15
    const int l15  = lane & 15;          // column within wave's 16-col tile
    const int col  = blockIdx.x * 64 + wave * 16 + l15;

    auto stage = [&](int buf, int qt) {
        const uint4* src = (const uint4*)(Qt + ((size_t)n * HW + qt * 16) * CQK); // 384 x 16B
        uint4* dst = (uint4*)sQ[buf];
        #pragma unroll
        for (int j = 0; j < 3; ++j) dst[threadIdx.x + j * 128] = src[threadIdx.x + j * 128];
        if (threadIdx.x < 64) {
            int c = threadIdx.x >> 4, r = threadIdx.x & 15;
            sCam[buf][c * 16 + r] = camn[((size_t)(n * 4 + c)) * HW + qt * 16 + r];
        }
    };

    // Hoist B operand (this wave's 16 K-columns, full 192-deep) into registers.
    // bf16 B, contiguous k order used here: lanes 0-15 col=l15 hold K=0..15, lanes 16-31 K=16..31.
    v16bf B[6];
    const unsigned short* kbase = Kt + ((size_t)n * HW + col) * CQK;
    #pragma unroll
    for (int cc = 0; cc < 6; ++cc) {
        union { uint4 q[2]; v16bf v; } U;
        const unsigned short* kp = kbase + cc * 32 + half * 16;
        U.q[0] = *(const uint4*)(kp);
        U.q[1] = *(const uint4*)(kp + 8);
        B[cc] = U.v;
    }

    float m = -1e30f, l = 0.f;
    float a0 = 0.f, a1 = 0.f, a2 = 0.f, a3 = 0.f;

    stage(0, 0);
    __syncthreads();

    for (int qt = 0; qt < 196; ++qt) {
        const int cur = qt & 1;

        // Overlap: stage next tile into the idle buffer while computing this one.
        if (qt + 1 < 196) stage(1 - cur, qt + 1);
        if (qt + 2 < 196)
            __builtin_prefetch(Qt + ((size_t)n * HW + (qt + 2) * 16) * CQK, 0, 1);

        // Load ALL six A chunks into distinct registers, then fence the
        // scheduler so they stay live and the WMMAs drain back-to-back.
        // bf16 A, contiguous k order used here: lane l15 row=l15, K = half*16 .. half*16+15 (matches B).
        v16bf A[6];
        const unsigned short* arow = sQ[cur] + l15 * CQK;
        #pragma unroll
        for (int cc = 0; cc < 6; ++cc) {
            union { uint4 q[2]; v16bf v; } U;
            const unsigned short* ap = arow + cc * 32 + half * 16;
            U.q[0] = *(const uint4*)(ap);
            U.q[1] = *(const uint4*)(ap + 8);
            A[cc] = U.v;
        }
        SCHED_FENCE();

        // S(16x16) = A(16x32) x B(32x16) over 6 K-chunks, f32 accum.
        v8f S = {};
        #pragma unroll
        for (int cc = 0; cc < 6; ++cc)
            S = __builtin_amdgcn_wmma_f32_16x16x32_bf16(
                    false, A[cc], false, B[cc], (short)0, S, false, false);
        SCHED_FENCE();

        // Online softmax over q for each column.
        float tmax = S[0];
        #pragma unroll
        for (int r = 1; r < 8; ++r) tmax = fmaxf(tmax, S[r]);
        tmax = fmaxf(tmax, __shfl_xor(tmax, 16, 32));
        float mnew  = fmaxf(m, tmax);
        float scale = __expf(m - mnew);
        l *= scale; a0 *= scale; a1 *= scale; a2 *= scale; a3 *= scale;

        float ps = 0.f, p0 = 0.f, p1 = 0.f, p2 = 0.f, p3 = 0.f;
        #pragma unroll
        for (int r = 0; r < 8; ++r) {
            int row = half * 8 + r;             // C/D row this VGPR holds
            float e = __expf(S[r] - mnew);
            ps += e;
            p0 += e * sCam[cur][0 * 16 + row];
            p1 += e * sCam[cur][1 * 16 + row];
            p2 += e * sCam[cur][2 * 16 + row];
            p3 += e * sCam[cur][3 * 16 + row];
        }
        ps += __shfl_xor(ps, 16, 32);
        p0 += __shfl_xor(p0, 16, 32);
        p1 += __shfl_xor(p1, 16, 32);
        p2 += __shfl_xor(p2, 16, 32);
        p3 += __shfl_xor(p3, 16, 32);
        l += ps; a0 += p0; a1 += p1; a2 += p2; a3 += p3;
        m = mnew;

        __syncthreads();   // next-tile staging complete; safe to rotate buffers
    }

    if (half == 0) {
        float inv = 1.f / l;
        size_t ob = (size_t)n * 4 * HW + col;
        out[ob + 0 * HW] = a0 * inv;
        out[ob + 1 * HW] = a1 * inv;
        out[ob + 2 * HW] = a2 * inv;
        out[ob + 3 * HW] = a3 * inv;
    }
}

extern "C" void kernel_launch(void* const* d_in, const int* in_sizes, int n_in,
                              void* d_out, int out_size, void* d_ws, size_t ws_size,
                              hipStream_t stream) {
    const float* x     = (const float*)d_in[0];
    /* x1 (d_in[1]) is unused by the reference */
    const float* x2    = (const float*)d_in[2];
    const float* deep3 = (const float*)d_in[3];
    const float* _4    = (const float*)d_in[4];
    const float* fc8_w = (const float*)d_in[5];
    const float* f83_w = (const float*)d_in[6];
    const float* f84_w = (const float*)d_in[7];
    const float* f91_w = (const float*)d_in[8];
    const float* f92_w = (const float*)d_in[9];
    float* out = (float*)d_out;

    char* ws = (char*)d_ws;
    float* cam  = (float*)(ws + 0);              //  8*4*HW f32   = 401,408 B
    float* camn = (float*)(ws + 401408);         //  8*4*HW f32   = 401,408 B
    float* f    = (float*)(ws + 802816);         //  8*195*HW f32 = 19,568,640 B
    char*  big  = ws + 20371456;                 //  x2r region, later reused for Qt/Kt
    float* x2r  = (float*)big;                   //  8*128*HW f32 = 12,845,056 B (dead after f83)
    unsigned short* Qt = (unsigned short*)big;              // 8*HW*192 bf16 = 9,633,792 B
    unsigned short* Kt = (unsigned short*)(big + 9633792);  // 8*HW*192 bf16 = 9,633,792 B

    cam_kernel      <<<dim3(8 * 4 * HW / 256),   256, 0, stream>>>(fc8_w, _4, cam);
    norm_kernel     <<<dim3(8),                  256, 0, stream>>>(cam, camn);
    resize_x_kernel <<<dim3(8 * 3 * HW / 256),   256, 0, stream>>>(x, f);
    resize_x2_kernel<<<dim3(8 * 128 * HW / 256), 256, 0, stream>>>(x2, x2r);
    f83_kernel      <<<dim3(8 * 64 * HW / 256),  256, 0, stream>>>(f83_w, x2r, f);
    f84_kernel      <<<dim3(8 * 128 * HW / 256), 256, 0, stream>>>(f84_w, deep3, f);
    qk_kernel       <<<dim3(8 * HW * CQK / 256), 256, 0, stream>>>(f91_w, f92_w, f, Qt, Kt);
    attn_kernel     <<<dim3(49, 8),              128, 0, stream>>>(Qt, Kt, camn, out);
}
